// MultiHeadSelfAttention_84421877170302
// MI455X (gfx1250) — matrix operand order repaired and checked by reading, not compiled, not run on hardware
//
#include <hip/hip_runtime.h>

typedef __attribute__((ext_vector_type(16))) _Float16 v16h;
typedef __attribute__((ext_vector_type(8)))  float    v8f;
typedef __attribute__((ext_vector_type(4)))  int      i4;

#define DEV __device__ __forceinline__

constexpr int Bb = 4, T = 2048, D = 1024, H = 16, DH = 64;
constexpr float SCALE = 0.125f;   // 1/sqrt(64)

// CDNA5 async global->LDS copy path (probe-guarded; falls back to b128 reg copy)
#if defined(__gfx1250__) && __has_builtin(__builtin_amdgcn_global_load_async_to_lds_b128) && \
    __has_builtin(__builtin_amdgcn_s_wait_asynccnt)
#define USE_ASYNC_LDS 1
typedef __attribute__((address_space(1))) i4* g_i4p;   // global int4*
typedef __attribute__((address_space(3))) i4* l_i4p;   // LDS int4*
#else
#define USE_ASYNC_LDS 0
#endif

DEV int lane_id() { return (int)(threadIdx.x & 31); }

// D = A(16x32 f16) * B(32x16 f16) + C(16x16 f32), wave32 WMMA
DEV v8f wmma(v16h a, v16h b, v8f c) {
  return __builtin_amdgcn_wmma_f32_16x16x32_f16(
      /*neg_a=*/false, a, /*neg_b=*/false, b,
      /*c_mod=*/(short)0, c, /*reuse_a=*/false, /*reuse_b=*/false);
}

// ---- Fragment loaders (wave32), contiguous k order, same for A and B ----
// A 16x32 (MxK), row-major source, row stride ld (elements).
// lane l: m = l&15, g = l>>4; VGPR pair p: K = 16g + 2p + {0,1}
DEV v16h load_a_f16(const _Float16* __restrict__ src, int ld) {
  int g = lane_id() >> 4, m = lane_id() & 15;
  const _Float16* r = src + (size_t)m * ld;
  v16h a;
#pragma unroll
  for (int p = 0; p < 8; ++p) {
    int k = 16 * g + 2 * p;
    a[2 * p]     = r[k];
    a[2 * p + 1] = r[k + 1];
  }
  return a;
}
DEV v16h load_a_f32(const float* __restrict__ src, int ld) {
  int g = lane_id() >> 4, m = lane_id() & 15;
  const float* r = src + (size_t)m * ld;
  v16h a;
#pragma unroll
  for (int p = 0; p < 8; ++p) {
    int k = 16 * g + 2 * p;
    a[2 * p]     = (_Float16)r[k];
    a[2 * p + 1] = (_Float16)r[k + 1];
  }
  return a;
}
// B 32x16 (KxN), row-major (K-major) source, row stride ld.
// lane l: n = l&15, g = l>>4; VGPR pair p: K = 16g + 2p + {0,1}
DEV v16h load_b_f16(const _Float16* src, int ld) {
  int g = lane_id() >> 4, n = lane_id() & 15;
  v16h b;
#pragma unroll
  for (int p = 0; p < 8; ++p) {
    int k = 16 * g + 2 * p;
    b[2 * p]     = src[(size_t)k * ld + n];
    b[2 * p + 1] = src[(size_t)(k + 1) * ld + n];
  }
  return b;
}
// Transposed B: B[k][n] = W[n*ld + k]  (for y = x @ W^T style weights)
DEV v16h load_bt_f16(const _Float16* W, int ld) {
  int g = lane_id() >> 4, n = lane_id() & 15;
  const _Float16* r = W + (size_t)n * ld;
  v16h b;
#pragma unroll
  for (int p = 0; p < 8; ++p) {
    int k = 16 * g + 2 * p;
    b[2 * p]     = r[k];
    b[2 * p + 1] = r[k + 1];
  }
  return b;
}
DEV v16h load_bt_f32(const float* __restrict__ W, int ld) {
  int g = lane_id() >> 4, n = lane_id() & 15;
  const float* r = W + (size_t)n * ld;
  v16h b;
#pragma unroll
  for (int p = 0; p < 8; ++p) {
    int k = 16 * g + 2 * p;
    b[2 * p]     = (_Float16)r[k];
    b[2 * p + 1] = (_Float16)r[k + 1];
  }
  return b;
}
// C/D 16x16 f32: VGPR v holds row (v + 8g), lane&15 = column.
DEV void store_c_f16(_Float16* __restrict__ dst, int ld, v8f c) {
  int g = lane_id() >> 4, n = lane_id() & 15;
#pragma unroll
  for (int v = 0; v < 8; ++v) dst[(size_t)(v + 8 * g) * ld + n] = (_Float16)c[v];
}
DEV void store_c_f32(float* __restrict__ dst, int ld, v8f c) {
  int g = lane_id() >> 4, n = lane_id() & 15;
#pragma unroll
  for (int v = 0; v < 8; ++v) dst[(size_t)(v + 8 * g) * ld + n] = c[v];
}

// ---- Cooperative staging of one contiguous 32x64 f16 tile (4 KB) into LDS ----
// 128 threads x 32 B each. Async path uses CDNA5 global_load_async_to_lds_b128
// (no VGPR round-trip, tracked by ASYNCcnt); fallback is a b128 register copy.
DEV void stage_tile(const _Float16* g, _Float16* l) {
  int t = (int)threadIdx.x;  // 0..127
#if USE_ASYNC_LDS
  const char* gc = (const char*)g;
  char* lc = (char*)l;
#pragma unroll
  for (int i = 0; i < 2; ++i) {
    __builtin_amdgcn_global_load_async_to_lds_b128(
        (g_i4p)(gc + (size_t)(t + 128 * i) * 16),
        (l_i4p)(lc + (size_t)(t + 128 * i) * 16),
        /*offset=*/0, /*cpol=*/0);
  }
#else
  const uint4* gs = (const uint4*)g;
  uint4* ls = (uint4*)l;
  ls[t] = gs[t];
  ls[t + 128] = gs[t + 128];
#endif
}
DEV void wait_stage() {
#if USE_ASYNC_LDS
  __builtin_amdgcn_s_wait_asynccnt(0);
#endif
}

// ---------------- Kernel 1: fused QKV head projections ----------------
__global__ __launch_bounds__(256) void qkv_kernel(
    const float* __restrict__ x, const float* __restrict__ Wq,
    const float* __restrict__ Wk, const float* __restrict__ Wv,
    _Float16* __restrict__ q, _Float16* __restrict__ k, _Float16* __restrict__ v) {
  int wid = blockIdx.x * 8 + (threadIdx.x >> 5);
  int tt = wid & (T / 16 - 1);
  int h  = (wid >> 7) & (H - 1);
  int b  = wid >> 11;

  const float* xa = x + ((size_t)b * T + (size_t)tt * 16) * D + h * DH;
  v16h a0 = load_a_f32(xa, D);       // K = 0..31 of DH
  v16h a1 = load_a_f32(xa + 32, D);  // K = 32..63

  const float* Ws[3] = {Wq + (size_t)h * DH * DH, Wk + (size_t)h * DH * DH,
                        Wv + (size_t)h * DH * DH};
  size_t obase = ((size_t)(b * H + h) * T + (size_t)tt * 16) * DH;
  _Float16* Os[3] = {q + obase, k + obase, v + obase};

#pragma unroll
  for (int w = 0; w < 3; ++w) {
#pragma unroll
    for (int nt = 0; nt < 4; ++nt) {
      const float* Wb = Ws[w] + (size_t)(nt * 16) * DH;  // rows n, cols k
      v8f c = {};
      c = wmma(a0, load_bt_f32(Wb, DH), c);
      c = wmma(a1, load_bt_f32(Wb + 32, DH), c);
      store_c_f16(Os[w] + nt * 16, DH, c);
    }
  }
}

// ---------------- Kernel 2: flash attention (online softmax) ----------------
// One block (4 waves) -> 64 query rows of one (b,h). K/V tiles staged in LDS
// once per block (double buffered), shared by all 4 waves.
__global__ __launch_bounds__(128) void flash_kernel(
    const _Float16* __restrict__ q, const _Float16* __restrict__ kk,
    const _Float16* __restrict__ vv, _Float16* __restrict__ attn) {
  __shared__ _Float16 kbuf[2][32 * 64];   // 2 x 4 KB
  __shared__ _Float16 vbuf[2][32 * 64];   // 2 x 4 KB
  __shared__ _Float16 pbuf[4][16 * 32];   // per-wave P staging (4 x 1 KB)

  int wv = threadIdx.x >> 5;
  int qb = blockIdx.x & (T / 64 - 1);        // 32 query blocks of 64
  int h  = (blockIdx.x >> 5) & (H - 1);
  int b  = blockIdx.x >> 9;
  int tt = qb * 4 + wv;                      // this wave's 16-row q tile

  size_t hb = (size_t)(b * H + h) * T * DH;
  const _Float16* Qb = q + hb + (size_t)tt * 16 * DH;
  const _Float16* Kb = kk + hb;
  const _Float16* Vb = vv + hb;

  v16h aq0 = load_a_f16(Qb, DH);
  v16h aq1 = load_a_f16(Qb + 32, DH);

  v8f o0 = {}, o1 = {}, o2 = {}, o3 = {};
  float mrow[8], lrow[8];
#pragma unroll
  for (int r = 0; r < 8; ++r) { mrow[r] = -1e30f; lrow[r] = 0.0f; }

  _Float16* pb = pbuf[wv];
  int g = lane_id() >> 4, c = lane_id() & 15;

  stage_tile(Kb, kbuf[0]);
  stage_tile(Vb, vbuf[0]);
  int buf = 0;

  for (int j = 0; j < T; j += 32) {
    wait_stage();          // this wave's async copies into kbuf/vbuf[buf] done
    __syncthreads();       // all waves' copies visible; prev readers drained
    if (j + 32 < T) {      // overlap: fill the other buffer during compute
      stage_tile(Kb + (size_t)(j + 32) * DH, kbuf[buf ^ 1]);
      stage_tile(Vb + (size_t)(j + 32) * DH, vbuf[buf ^ 1]);
    }
    const _Float16* Kl = kbuf[buf];
    const _Float16* Vl = vbuf[buf];

    // S tiles for keys [j, j+16) and [j+16, j+32); contraction over DH=64
    v8f s0 = {}, s1 = {};
    s0 = wmma(aq0, load_bt_f16(Kl, DH), s0);
    s0 = wmma(aq1, load_bt_f16(Kl + 32, DH), s0);
    s1 = wmma(aq0, load_bt_f16(Kl + 16 * DH, DH), s1);
    s1 = wmma(aq1, load_bt_f16(Kl + 16 * DH + 32, DH), s1);

#pragma unroll
    for (int r = 0; r < 8; ++r) {  // row = r + 8g, cols across 16 lanes of half
      float x0 = s0[r] * SCALE, x1 = s1[r] * SCALE;
      float mx = fmaxf(x0, x1);
      mx = fmaxf(mx, __shfl_xor(mx, 1, 16));
      mx = fmaxf(mx, __shfl_xor(mx, 2, 16));
      mx = fmaxf(mx, __shfl_xor(mx, 4, 16));
      mx = fmaxf(mx, __shfl_xor(mx, 8, 16));
      float mnew  = fmaxf(mrow[r], mx);
      float alpha = __expf(mrow[r] - mnew);
      float p0 = __expf(x0 - mnew), p1 = __expf(x1 - mnew);
      float rs = p0 + p1;
      rs += __shfl_xor(rs, 1, 16);
      rs += __shfl_xor(rs, 2, 16);
      rs += __shfl_xor(rs, 4, 16);
      rs += __shfl_xor(rs, 8, 16);
      lrow[r] = lrow[r] * alpha + rs;
      mrow[r] = mnew;
      o0[r] *= alpha; o1[r] *= alpha; o2[r] *= alpha; o3[r] *= alpha;
      int row = r + 8 * g;
      pb[row * 32 + c]      = (_Float16)p0;   // K index = col (keys j..j+15)
      pb[row * 32 + 16 + c] = (_Float16)p1;   // K index = 16+col
    }
    // LDS ops are in-order within a wave: P stores above precede this load
    v16h ap = load_a_f16(pb, 32);
    o0 = wmma(ap, load_b_f16(Vl + 0,  DH), o0);
    o1 = wmma(ap, load_b_f16(Vl + 16, DH), o1);
    o2 = wmma(ap, load_b_f16(Vl + 32, DH), o2);
    o3 = wmma(ap, load_b_f16(Vl + 48, DH), o3);

    __syncthreads();       // everyone done reading buf before it is refilled
    buf ^= 1;
  }

  _Float16* ob = attn + ((size_t)b * T + (size_t)tt * 16) * D + h * DH;
#pragma unroll
  for (int r = 0; r < 8; ++r) {
    float inv = 1.0f / lrow[r];
    size_t ro = (size_t)(r + 8 * g) * D;
    ob[ro + c]      = (_Float16)(o0[r] * inv);
    ob[ro + 16 + c] = (_Float16)(o1[r] * inv);
    ob[ro + 32 + c] = (_Float16)(o2[r] * inv);
    ob[ro + 48 + c] = (_Float16)(o3[r] * inv);
  }
}

// ---------------- Kernel 3: output projection  y = attn @ Wout^T ----------------
__global__ __launch_bounds__(256) void out_proj_kernel(
    const _Float16* __restrict__ a, const float* __restrict__ Wout,
    float* __restrict__ y) {
  int wid = blockIdx.x * 8 + (threadIdx.x >> 5);
  int mt = wid & (Bb * T / 16 - 1);  // 512 row tiles
  int ng = wid >> 9;                 // 16 col groups of 64
  const _Float16* A = a + (size_t)mt * 16 * D;
  v8f c0 = {}, c1 = {}, c2 = {}, c3 = {};
  for (int kc = 0; kc < D; kc += 32) {
    v16h af = load_a_f16(A + kc, D);
    const float* Wb = Wout + (size_t)(ng * 64) * D + kc;  // rows n, cols k
    c0 = wmma(af, load_bt_f32(Wb + (size_t)0 * 16 * D, D), c0);
    c1 = wmma(af, load_bt_f32(Wb + (size_t)1 * 16 * D, D), c1);
    c2 = wmma(af, load_bt_f32(Wb + (size_t)2 * 16 * D, D), c2);
    c3 = wmma(af, load_bt_f32(Wb + (size_t)3 * 16 * D, D), c3);
  }
  float* Y = y + (size_t)mt * 16 * D + ng * 64;
  store_c_f32(Y + 0,  D, c0);
  store_c_f32(Y + 16, D, c1);
  store_c_f32(Y + 32, D, c2);
  store_c_f32(Y + 48, D, c3);
}

extern "C" void kernel_launch(void* const* d_in, const int* in_sizes, int n_in,
                              void* d_out, int out_size, void* d_ws, size_t ws_size,
                              hipStream_t stream) {
  (void)in_sizes; (void)n_in; (void)out_size; (void)ws_size;
  const float* x    = (const float*)d_in[0];
  const float* Wq   = (const float*)d_in[1];
  const float* Wk   = (const float*)d_in[2];
  const float* Wv   = (const float*)d_in[3];
  const float* Wout = (const float*)d_in[4];

  const size_t n = (size_t)Bb * H * T * DH;  // 8,388,608 elements
  _Float16* q  = (_Float16*)d_ws;
  _Float16* k  = q + n;
  _Float16* v  = k + n;
  _Float16* at = v + n;  // (B*T, D) f16 attention output

  const int waves = Bb * H * (T / 16);       // 8192
  qkv_kernel<<<waves / 8, 256, 0, stream>>>(x, Wq, Wk, Wv, q, k, v);
  flash_kernel<<<Bb * H * (T / 64), 128, 0, stream>>>(q, k, v, at);
  out_proj_kernel<<<(Bb * T / 16) * (D / 64) / 8, 256, 0, stream>>>(at, Wout, (float*)d_out);
}
